// MLP2plus_49057116455327
// MI455X (gfx1250) — matrix operand order repaired and checked by reading, not compiled, not run on hardware
//
#include <hip/hip_runtime.h>
#include <hip/hip_bf16.h>
#include <math.h>

typedef _Float16 v4h  __attribute__((ext_vector_type(4)));
typedef _Float16 v8h  __attribute__((ext_vector_type(8)));
typedef _Float16 v16h __attribute__((ext_vector_type(16)));
typedef float    v8f  __attribute__((ext_vector_type(8)));

#define NEU   1024
#define BATCH 131072
#define HS    1032   // padded LDS row stride in halves: 516 dwords ≡ 4 mod 64 banks
#define MT    64     // batch rows per workgroup (4 m-fragments)

// ---------------------------------------------------------------- W2 -> f16
__global__ __launch_bounds__(256) void w2_to_f16(const float* __restrict__ w,
                                                 _Float16* __restrict__ o) {
    int i = (blockIdx.x * blockDim.x + threadIdx.x) * 4;
    float4 v = *(const float4*)(w + i);
    v4h h = { (_Float16)v.x, (_Float16)v.y, (_Float16)v.z, (_Float16)v.w };
    *(v4h*)(o + i) = h;
}

__device__ inline v16h pack16(v8h lo, v8h hi) {
    return __builtin_shufflevector(lo, hi, 0,1,2,3,4,5,6,7,8,9,10,11,12,13,14,15);
}

// ---------------------------------------------------------------- fused MLP
__global__ __launch_bounds__(256) void mlp_fused(
    const float* __restrict__ x,
    const float* __restrict__ Wx, const float* __restrict__ bx,
    const float* __restrict__ Wu, const float* __restrict__ bu,
    const float* __restrict__ W1, const float* __restrict__ b1,
    const _Float16* __restrict__ W2h, const float* __restrict__ b2,
    const float* __restrict__ W3, const float* __restrict__ b3,
    float* __restrict__ out)
{
    __shared__ float    l1s[MT][12];
    __shared__ _Float16 h1s[MT * HS];     // 132 KB, f16 activations
    __shared__ float    outAcc[MT][3];

    const int t    = threadIdx.x;
    const int row0 = blockIdx.x * MT;

    if (t < MT * 3) ((float*)outAcc)[t] = 0.f;

    // ---- stage 0: strided feature gather + 2x5 ReLU projections -> l1 (f32)
    if (t < MT) {
        float xr[25];
        const float* xp = x + (size_t)(row0 + t) * 25;
        #pragma unroll
        for (int c = 0; c < 25; ++c) xr[c] = xp[c];
        float gv[5][5];
        #pragma unroll
        for (int i = 0; i < 5; ++i) {
            gv[0][i] = xr[3*i];        // vx
            gv[1][i] = xr[3*i + 1];    // vy
            gv[2][i] = xr[3*i + 2];    // r
            gv[3][i] = xr[15 + 2*i];   // th
            gv[4][i] = xr[16 + 2*i];   // st
        }
        #pragma unroll
        for (int g = 0; g < 5; ++g) {
            const float* Wg = (g < 3) ? Wx : Wu;
            const float* bg = (g < 3) ? bx : bu;
            #pragma unroll
            for (int o = 0; o < 2; ++o) {
                float a = bg[o];
                #pragma unroll
                for (int i = 0; i < 5; ++i) a += Wg[o*5 + i] * gv[g][i];
                l1s[t][g*2 + o] = a > 0.f ? a : 0.f;
            }
        }
    }
    __syncthreads();

    // ---- stage 1 (VALU, ~1% of FLOPs): h1 = relu(l1 @ W1^T + b1) -> LDS f16
    {
        float w1r[4][10], b1r[4];
        #pragma unroll
        for (int ni = 0; ni < 4; ++ni) {
            const int n = ni * 256 + t;
            b1r[ni] = b1[n];
            #pragma unroll
            for (int j = 0; j < 10; ++j) w1r[ni][j] = W1[n*10 + j];
        }
        #pragma unroll 4
        for (int m = 0; m < MT; ++m) {
            float lr[10];
            #pragma unroll
            for (int j = 0; j < 10; ++j) lr[j] = l1s[m][j];
            #pragma unroll
            for (int ni = 0; ni < 4; ++ni) {
                const int n = ni * 256 + t;
                float a = b1r[ni];
                #pragma unroll
                for (int j = 0; j < 10; ++j) a += w1r[ni][j] * lr[j];
                h1s[m * HS + n] = (_Float16)(a > 0.f ? a : 0.f);
            }
        }
    }
    __syncthreads();

    // ---- stage 2: h2 = relu(h1 @ W2^T + b2) via WMMA (4m x 4n register block)
    const int lane   = t & 31;
    const int q      = t >> 5;          // wave: owns n-tiles [q*8, q*8+8)
    const int hiHalf = (lane & 16) ? 1 : 0;
    const int rIdx   = lane & 15;
    const int aSel   = hiHalf ? 8 : 0;  // A layout: K {0..7,16..23}/{8..15,24..31}
    const int bSel   = hiHalf ? 8 : 0;  // B layout: K {0..7,16..23}/{8..15,24..31}

    const _Float16* aRow[4];
    #pragma unroll
    for (int m = 0; m < 4; ++m)
        aRow[m] = h1s + (size_t)(m * 16 + rIdx) * HS + aSel;

    float accO[4][8][3];                // fused W3 contraction accumulators
    #pragma unroll
    for (int m = 0; m < 4; ++m)
        #pragma unroll
        for (int v = 0; v < 8; ++v)
            { accO[m][v][0] = 0.f; accO[m][v][1] = 0.f; accO[m][v][2] = 0.f; }

    #pragma unroll 1
    for (int g = 0; g < 2; ++g) {
        const int nbase = (q * 8 + g * 4) * 16;
        const _Float16* bPtr[4];
        #pragma unroll
        for (int u = 0; u < 4; ++u)
            bPtr[u] = W2h + (size_t)(nbase + u * 16 + rIdx) * NEU + bSel;

        v8f acc[4][4] = {};

        #pragma unroll 1
        for (int kc = 0; kc < NEU; kc += 32) {
            v16h b[4];
            #pragma unroll
            for (int u = 0; u < 4; ++u)
                b[u] = pack16(*(const v8h*)(bPtr[u] + kc),
                              *(const v8h*)(bPtr[u] + kc + 16));
            #pragma unroll
            for (int m = 0; m < 4; ++m) {
                v16h a = pack16(*(const v8h*)(aRow[m] + kc),
                                *(const v8h*)(aRow[m] + kc + 16));
                #pragma unroll
                for (int u = 0; u < 4; ++u)
                    acc[m][u] = __builtin_amdgcn_wmma_f32_16x16x32_f16(
                        false, a, false, b[u], (short)0, acc[m][u], false, false);
            }
        }

        // bias + ReLU + fused contraction with the 3 W3 rows
        #pragma unroll
        for (int u = 0; u < 4; ++u) {
            const int n   = nbase + u * 16 + rIdx;
            const float bn  = b2[n];
            const float w30 = W3[n], w31 = W3[NEU + n], w32 = W3[2*NEU + n];
            #pragma unroll
            for (int m = 0; m < 4; ++m) {
                #pragma unroll
                for (int v = 0; v < 8; ++v) {
                    float h = acc[m][u][v] + bn;
                    h = h > 0.f ? h : 0.f;
                    accO[m][v][0] += h * w30;
                    accO[m][v][1] += h * w31;
                    accO[m][v][2] += h * w32;
                }
            }
        }
    }

    // reduce over the 16 N-columns held across each 16-lane half, combine waves
    #pragma unroll
    for (int m = 0; m < 4; ++m) {
        #pragma unroll
        for (int v = 0; v < 8; ++v) {
            #pragma unroll
            for (int j = 0; j < 3; ++j) {
                float s = accO[m][v][j];
                s += __shfl_xor(s, 1, 32);
                s += __shfl_xor(s, 2, 32);
                s += __shfl_xor(s, 4, 32);
                s += __shfl_xor(s, 8, 32);
                if (rIdx == 0)
                    atomicAdd(&outAcc[m * 16 + hiHalf * 8 + v][j], s);
            }
        }
    }
    __syncthreads();

    // ---- stage 3: sigmoid(acc + b3) -> out (BATCH,1,3)
    if (t < MT * 3) {
        const int m = t / 3, j = t % 3;
        const float v = outAcc[m][j] + b3[j];
        out[(size_t)(row0 + m) * 3 + j] = 1.f / (1.f + __expf(-v));
    }
}

// ---------------------------------------------------------------- launcher
extern "C" void kernel_launch(void* const* d_in, const int* in_sizes, int n_in,
                              void* d_out, int out_size, void* d_ws, size_t ws_size,
                              hipStream_t stream) {
    const float* x  = (const float*)d_in[0];
    const float* Wx = (const float*)d_in[1];
    const float* bx = (const float*)d_in[2];
    const float* Wu = (const float*)d_in[3];
    const float* bu = (const float*)d_in[4];
    const float* W1 = (const float*)d_in[5];
    const float* b1 = (const float*)d_in[6];
    const float* W2 = (const float*)d_in[7];
    const float* b2 = (const float*)d_in[8];
    const float* W3 = (const float*)d_in[9];
    const float* b3 = (const float*)d_in[10];
    float*    out  = (float*)d_out;
    _Float16* W2h  = (_Float16*)d_ws;     // 1024*1024*2 = 2 MB scratch

    hipLaunchKernelGGL(w2_to_f16, dim3((NEU * NEU / 4) / 256), dim3(256), 0, stream,
                       W2, W2h);
    hipLaunchKernelGGL(mlp_fused, dim3(BATCH / MT), dim3(256), 0, stream,
                       x, Wx, bx, Wu, bu, W1, b1, W2h, b2, W3, b3, out);
}
